// SparseMultiHeadAttention_63127429316731
// MI455X (gfx1250) — matrix operand order repaired and checked by reading, not compiled, not run on hardware
//
#include <hip/hip_runtime.h>
#include <hip/hip_bf16.h>

#define N_HEADS   16
#define N_ACT     4
#define DM        1024
#define DA        64
#define BATCH     4
#define SEQ       2048

typedef __attribute__((ext_vector_type(4)))  unsigned int v4u;
typedef __attribute__((ext_vector_type(8)))  float        v8f;
typedef __attribute__((ext_vector_type(16))) __bf16       v16bf;
typedef v4u __attribute__((may_alias)) v4u_ma;

#if defined(__AMDGCN__) && __has_builtin(__builtin_amdgcn_global_load_async_to_lds_b128) && __has_builtin(__builtin_amdgcn_s_wait_asynccnt)
#define USE_ASYNC_LDS 1
typedef int v4i_vs __attribute__((__vector_size__(4 * sizeof(int))));
typedef __attribute__((address_space(1))) v4i_vs as1_v4i;   // global int4
typedef __attribute__((address_space(3))) v4i_vs as3_v4i;   // LDS int4
#else
#define USE_ASYNC_LDS 0
#endif

union FragU { v4u u[2]; v16bf h; };

// ---------------- workspace layout (bytes) ----------------
static constexpr size_t OFF_IDX  = 0;
static constexpr size_t OFF_XP   = 256;                   // 16 MB  x A-frags
static constexpr size_t OFF_WP   = OFF_XP   + 16777216;   // 6 MB   W B-frags
static constexpr size_t OFF_QF   = OFF_WP   + 6291456;    // 8 MB
static constexpr size_t OFF_KF   = OFF_QF   + 8388608;    // 8 MB
static constexpr size_t OFF_VF   = OFF_KF   + 8388608;    // 8 MB
static constexpr size_t OFF_QP   = OFF_VF   + 8388608;    // 4 MB   Q A-frags
static constexpr size_t OFF_KP   = OFF_QP   + 4194304;    // 4 MB   K^T B-frags
static constexpr size_t OFF_VP   = OFF_KP   + 4194304;    // 4 MB   V B-frags
static constexpr size_t OFF_OACT = OFF_VP   + 4194304;    // 8 MB
static constexpr size_t OFF_OP   = OFF_OACT + 8388608;    // 4 MB   O A-frags
static constexpr size_t OFF_WOP  = OFF_OP   + 4194304;    // 0.5 MB Wo B-frags

// ---------------- helpers ----------------
static __device__ __forceinline__ unsigned int pack_bf16x2(float a, float b) {
  unsigned int ua = __builtin_bit_cast(unsigned int, a);
  unsigned int ub = __builtin_bit_cast(unsigned int, b);
  ua = (ua + 0x7FFFu + ((ua >> 16) & 1u)) >> 16;   // RNE
  ub = (ub + 0x7FFFu + ((ub >> 16) & 1u)) >> 16;
  return (ua & 0xFFFFu) | (ub << 16);
}
static __device__ __forceinline__ unsigned short f32_bf16(float a) {
  unsigned int ua = __builtin_bit_cast(unsigned int, a);
  return (unsigned short)((ua + 0x7FFFu + ((ua >> 16) & 1u)) >> 16);
}
static __device__ __forceinline__ v16bf load_frag(const unsigned int* base, int lane) {
  FragU f;
  const v4u_ma* p = (const v4u_ma*)(base + lane * 8);
  f.u[0] = p[0];
  f.u[1] = p[1];
  return f.h;
}
static __device__ __forceinline__ v8f wmma_bf16(v16bf a, v16bf b, v8f c) {
  return __builtin_amdgcn_wmma_f32_16x16x32_bf16(false, a, false, b, (short)0, c, false, false);
}
static __device__ __forceinline__ int a_koff(int j, int h) {   // A-frag K offset
  return (j << 1) + (h << 4);
}
static __device__ __forceinline__ v8f v8f_zero() {
  v8f z = {0.f, 0.f, 0.f, 0.f, 0.f, 0.f, 0.f, 0.f};
  return z;
}

// ---------------- router: top-4 heads of mean logits, sorted ----------------
__global__ __launch_bounds__(256) void router_kernel(
    const float* __restrict__ x, const float* __restrict__ Wr,
    const float* __restrict__ br, int* __restrict__ idx_out) {
  __shared__ float xbar[DM];
  __shared__ float logits[N_HEADS];
  const int b = blockIdx.x, t = threadIdx.x;
  for (int c = t; c < DM; c += 256) {
    float acc = 0.f;
    const float* p = x + (size_t)b * SEQ * DM + c;
    for (int s = 0; s < SEQ; ++s) acc += p[(size_t)s * DM];
    xbar[c] = acc;                       // sum; top-k invariant to /S
  }
  __syncthreads();
  if (t < N_HEADS) {
    float acc = br[t] * (float)SEQ;      // match the sum scale
    for (int d = 0; d < DM; ++d) acc += xbar[d] * Wr[d * N_HEADS + t];
    logits[t] = acc;
  }
  __syncthreads();
  if (t == 0) {
    float v[N_HEADS];
    for (int i = 0; i < N_HEADS; ++i) v[i] = logits[i];
    int sel[N_ACT];
    for (int a = 0; a < N_ACT; ++a) {
      int best = 0; float bvv = v[0];
      for (int i = 1; i < N_HEADS; ++i) if (v[i] > bvv) { bvv = v[i]; best = i; }
      sel[a] = best; v[best] = -__builtin_inff();
    }
    for (int i = 1; i < N_ACT; ++i) {    // sort ascending
      int key = sel[i], jj = i - 1;
      while (jj >= 0 && sel[jj] > key) { sel[jj + 1] = sel[jj]; --jj; }
      sel[jj + 1] = key;
    }
    for (int a = 0; a < N_ACT; ++a) idx_out[b * N_ACT + a] = sel[a];
  }
}

// ---------------- pack x -> bf16 A-fragments ----------------
__global__ __launch_bounds__(256) void pack_x_kernel(
    const float* __restrict__ x, unsigned int* __restrict__ Xp) {
  unsigned id = blockIdx.x * 256u + threadIdx.x;        // 4,194,304 dwords
  const int j = id & 7, lane = (id >> 3) & 31, kt = (id >> 8) & 31;
  const int mt = (id >> 13) & 127, b = id >> 20;
  const int m = mt * 16 + (lane & 15), h = lane >> 4;
  const int k = kt * 32 + a_koff(j, h);
  const float* src = x + (((size_t)b * SEQ + m) << 10) + k;
  Xp[id] = pack_bf16x2(src[0], src[1]);
}

// ---------------- gather active-head W columns -> bf16 B-fragments ----------------
__global__ __launch_bounds__(256) void pack_w_kernel(
    const float* __restrict__ Wq, const float* __restrict__ Wk,
    const float* __restrict__ Wv, const int* __restrict__ idx,
    unsigned int* __restrict__ Wp) {
  unsigned id = blockIdx.x * 256u + threadIdx.x;        // 1,572,864 dwords
  const int j = id & 7, lane = (id >> 3) & 31, nt = (id >> 8) & 3;
  const int kt = (id >> 10) & 31, ba = (id >> 15) & 15, qkv = id >> 19;
  const int head = idx[ba];
  const int n = lane & 15, h = lane >> 4;
  const int krow = kt * 32 + 2 * j + 16 * h;
  const int col = head * DA + nt * 16 + n;
  const float* W = (qkv == 0) ? Wq : (qkv == 1) ? Wk : Wv;
  Wp[id] = pack_bf16x2(W[(size_t)krow * DM + col], W[(size_t)(krow + 1) * DM + col]);
}

// ---------------- QKV projection GEMM, 2x2 register-blocked ----------------
__global__ __launch_bounds__(256) void proj_gemm_kernel(
    const unsigned int* __restrict__ Xp, const unsigned int* __restrict__ Wp,
    const int* __restrict__ idx,
    const float* __restrict__ bq, const float* __restrict__ bk, const float* __restrict__ bv,
    float* __restrict__ Qf, float* __restrict__ Kf, float* __restrict__ Vf) {
  const int lane = threadIdx.x & 31, wave = threadIdx.x >> 5;
  const int qkv = blockIdx.y >> 4, ba = blockIdx.y & 15;
  const int b = ba >> 2;
  const int t = blockIdx.x * 8 + wave;                  // 0..127
  const int mt2 = t >> 1, nt2 = t & 1;                  // 64 x 2 blocks of 32x32
  const unsigned int* A0 = Xp + ((size_t)(b * 128 + 2 * mt2 + 0) * 32) * 256;
  const unsigned int* A1 = Xp + ((size_t)(b * 128 + 2 * mt2 + 1) * 32) * 256;
  const unsigned int* Bb = Wp + (((size_t)(qkv * 16 + ba) * 32) * 4 + 2 * nt2) * 256;
  v8f acc[2][2];
  #pragma unroll
  for (int i = 0; i < 2; ++i)
    #pragma unroll
    for (int jj = 0; jj < 2; ++jj) acc[i][jj] = v8f_zero();
  for (int kt = 0; kt < 32; ++kt) {
    __builtin_prefetch((const void*)(A0 + (kt + 1) * 256 + lane * 8), 0, 3);
    const v16bf a0 = load_frag(A0 + (size_t)kt * 256, lane);
    const v16bf a1 = load_frag(A1 + (size_t)kt * 256, lane);
    const v16bf b0 = load_frag(Bb + (size_t)kt * 4 * 256, lane);
    const v16bf b1 = load_frag(Bb + (size_t)kt * 4 * 256 + 256, lane);
    acc[0][0] = wmma_bf16(a0, b0, acc[0][0]);
    acc[0][1] = wmma_bf16(a0, b1, acc[0][1]);
    acc[1][0] = wmma_bf16(a1, b0, acc[1][0]);
    acc[1][1] = wmma_bf16(a1, b1, acc[1][1]);
  }
  const int n = lane & 15, h = lane >> 4;
  const int head = idx[ba];
  const float* bias = (qkv == 0) ? bq : (qkv == 1) ? bk : bv;
  float* O = ((qkv == 0) ? Qf : (qkv == 1) ? Kf : Vf) + (size_t)ba * SEQ * DA;
  #pragma unroll
  for (int i = 0; i < 2; ++i) {
    #pragma unroll
    for (int jj = 0; jj < 2; ++jj) {
      const int nt = 2 * nt2 + jj;
      const float bvv = bias[head * DA + nt * 16 + n];
      #pragma unroll
      for (int j = 0; j < 8; ++j) {
        const int r = (2 * mt2 + i) * 16 + j + 8 * h;
        O[(size_t)r * DA + nt * 16 + n] = acc[i][jj][j] + bvv;
      }
    }
  }
}

// ---------------- repack kernels ----------------
__global__ __launch_bounds__(256) void repack_q_kernel(
    const float* __restrict__ Qf, unsigned int* __restrict__ Qp) {
  unsigned id = blockIdx.x * 256u + threadIdx.x;
  const int j = id & 7, lane = (id >> 3) & 31, kt = (id >> 8) & 1;
  const int mt = (id >> 9) & 127, ba = id >> 16;
  const int m = mt * 16 + (lane & 15), h = lane >> 4;
  const int k = kt * 32 + a_koff(j, h);
  const float* s = Qf + ((size_t)ba * SEQ + m) * DA + k;
  Qp[id] = pack_bf16x2(s[0], s[1]);
}
__global__ __launch_bounds__(256) void repack_k_kernel(
    const float* __restrict__ Kf, unsigned int* __restrict__ Kp) {
  unsigned id = blockIdx.x * 256u + threadIdx.x;
  const int j = id & 7, lane = (id >> 3) & 31, nt = (id >> 8) & 127;
  const int kt = (id >> 15) & 1, ba = id >> 16;
  const int n = lane & 15, h = lane >> 4;
  const int d = kt * 32 + 2 * j + 16 * h;               // kdim of S=QK^T is d
  const int skey = nt * 16 + n;
  const float* s = Kf + ((size_t)ba * SEQ + skey) * DA + d;
  Kp[id] = pack_bf16x2(s[0], s[1]);
}
__global__ __launch_bounds__(256) void repack_v_kernel(
    const float* __restrict__ Vf, unsigned int* __restrict__ Vp) {
  unsigned id = blockIdx.x * 256u + threadIdx.x;
  const int j = id & 7, lane = (id >> 3) & 31, nt = (id >> 8) & 3;
  const int kt = (id >> 10) & 63, ba = id >> 16;
  const int n = lane & 15, h = lane >> 4;
  const int skey = kt * 32 + 2 * j + 16 * h;            // kdim of P@V is s_key
  const int d = nt * 16 + n;
  const float* s = Vf + ((size_t)ba * SEQ + skey) * DA + d;
  Vp[id] = pack_bf16x2(s[0], s[DA]);
}
__global__ __launch_bounds__(256) void repack_o_kernel(
    const float* __restrict__ Oact, unsigned int* __restrict__ Op) {
  unsigned id = blockIdx.x * 256u + threadIdx.x;
  const int j = id & 7, lane = (id >> 3) & 31, kt = (id >> 8) & 7;
  const int mt = (id >> 11) & 127, b = id >> 18;
  const int m = mt * 16 + (lane & 15), h = lane >> 4;
  const int k = kt * 32 + a_koff(j, h);                 // 0..255 = a*64+d, even
  const int a = k >> 6, d = k & 63;
  const float* s = Oact + ((size_t)(b * N_ACT + a) * SEQ + m) * DA + d;
  Op[id] = pack_bf16x2(s[0], s[1]);
}
__global__ __launch_bounds__(256) void pack_wo_kernel(
    const float* __restrict__ Wo, unsigned int* __restrict__ WoP) {
  unsigned id = blockIdx.x * 256u + threadIdx.x;
  const int j = id & 7, lane = (id >> 3) & 31, nt = (id >> 8) & 63, kt = id >> 14;
  const int n = lane & 15, h = lane >> 4;
  const int krow = kt * 32 + 2 * j + 16 * h;
  const int col = nt * 16 + n;
  WoP[id] = pack_bf16x2(Wo[(size_t)krow * DM + col], Wo[(size_t)(krow + 1) * DM + col]);
}

// ---------------- flash attention ----------------
// Block = 8 waves sharing (b,head). K/V fragments staged into LDS via async
// global->LDS copies (ASYNCcnt), triple-buffered with one barrier per step.
// Each wave owns 32 query rows (2 M-tiles); 16 WMMA per 32-key step.
__global__ __launch_bounds__(256) void attn_kernel(
    const unsigned int* __restrict__ Qp, const unsigned int* __restrict__ Kp,
    const unsigned int* __restrict__ Vp, float* __restrict__ Oact) {
  __shared__ __align__(16) unsigned int   kvbuf[3][8 * 256];     // 3 x 8KB staged K/V frags
  __shared__ __align__(16) unsigned short plds[8][2][16 * 32];   // per-wave P tiles
  const int tid = threadIdx.x;
  const int lane = tid & 31, wave = tid >> 5;
  const int ba = blockIdx.y;
  const int mt0 = (blockIdx.x * 8 + wave) * 2;          // 2 row-tiles per wave
  const int n = lane & 15, h = lane >> 4, m = lane & 15;

  const unsigned int* Kb = Kp + (size_t)ba * 2 * 128 * 256;
  const unsigned int* Vb = Vp + (size_t)ba * 64 * 4 * 256;

  v16bf qf[2][2];
  #pragma unroll
  for (int mi = 0; mi < 2; ++mi)
    #pragma unroll
    for (int kt = 0; kt < 2; ++kt)
      qf[mi][kt] = load_frag(Qp + ((size_t)(ba * 128 + mt0 + mi) * 2 + kt) * 256, lane);

  // staging role: 8 chunks of 1KB per step (K: ntsel*2+ktd for chunks 0-3, V: 4-7)
  const int chunk = tid >> 5, inner = tid & 31;
  const unsigned int* gbase;                            // per-thread 32B source
  if (chunk < 4) gbase = Kb + ((size_t)(chunk & 1) * 128 + (chunk >> 1)) * 256 + inner * 8;
  else           gbase = Vb + ((size_t)(chunk - 4)) * 256 + inner * 8;
  const size_t gstep = (chunk < 4) ? 512 : 1024;        // dwords advanced per step

  auto stage = [&](int it, int bufidx) {
    const unsigned int* g = gbase + (size_t)it * gstep;
    unsigned int* l = &kvbuf[bufidx][chunk * 256 + inner * 8];
#if USE_ASYNC_LDS
    __builtin_amdgcn_global_load_async_to_lds_b128(
        (as1_v4i*)(uintptr_t)g, (as3_v4i*)(uintptr_t)l, 0, 0);
    __builtin_amdgcn_global_load_async_to_lds_b128(
        (as1_v4i*)(uintptr_t)g, (as3_v4i*)(uintptr_t)l, 16, 0);
#else
    const v4u_ma* s = (const v4u_ma*)g;
    v4u_ma a = s[0], bb = s[1];
    v4u_ma* d = (v4u_ma*)l;
    d[0] = a; d[1] = bb;
#endif
  };

  v8f o[2][4];
  float mrun[2][8], lrun[2][8];
  #pragma unroll
  for (int mi = 0; mi < 2; ++mi) {
    #pragma unroll
    for (int t = 0; t < 4; ++t) o[mi][t] = v8f_zero();
    #pragma unroll
    for (int j = 0; j < 8; ++j) { mrun[mi][j] = -__builtin_inff(); lrun[mi][j] = 0.f; }
  }

  stage(0, 0);
  for (int it = 0; it < 64; ++it) {                     // 32 keys per step
    if (it + 1 < 64) {
      stage(it + 1, (it + 1) % 3);                      // overlap copy with compute
#if USE_ASYNC_LDS
      __builtin_amdgcn_s_wait_asynccnt(2);              // this step's batch done
#endif
    } else {
#if USE_ASYNC_LDS
      __builtin_amdgcn_s_wait_asynccnt(0);
#endif
    }
    __syncthreads();                                    // all chunks visible
    const unsigned int* buf = kvbuf[it % 3];

    v16bf kf[2][2];                                     // [ntsel][ktd]
    #pragma unroll
    for (int ns = 0; ns < 2; ++ns)
      #pragma unroll
      for (int kd = 0; kd < 2; ++kd)
        kf[ns][kd] = load_frag(buf + (ns * 2 + kd) * 256, lane);

    #pragma unroll
    for (int mi = 0; mi < 2; ++mi) {
      v8f s0 = wmma_bf16(qf[mi][1], kf[0][1], wmma_bf16(qf[mi][0], kf[0][0], v8f_zero()));
      v8f s1 = wmma_bf16(qf[mi][1], kf[1][1], wmma_bf16(qf[mi][0], kf[1][0], v8f_zero()));
      float p0[8], p1[8];
      #pragma unroll
      for (int j = 0; j < 8; ++j) {                     // online softmax per row
        float a0 = s0[j] * 0.125f, a1 = s1[j] * 0.125f; // 1/sqrt(64)
        float mx = fmaxf(a0, a1);
        #pragma unroll
        for (int off = 1; off < 16; off <<= 1) mx = fmaxf(mx, __shfl_xor(mx, off, 32));
        float mnew = fmaxf(mrun[mi][j], mx);
        float alpha = __expf(mrun[mi][j] - mnew);
        float e0 = __expf(a0 - mnew), e1 = __expf(a1 - mnew);
        float rs = e0 + e1;
        #pragma unroll
        for (int off = 1; off < 16; off <<= 1) rs += __shfl_xor(rs, off, 32);
        lrun[mi][j] = lrun[mi][j] * alpha + rs;
        mrun[mi][j] = mnew;
        p0[j] = e0; p1[j] = e1;
        #pragma unroll
        for (int t = 0; t < 4; ++t) o[mi][t][j] *= alpha;
      }
      unsigned short* L = plds[wave][mi];
      #pragma unroll
      for (int j = 0; j < 8; ++j) {                     // C-layout -> LDS (row, k)
        const int r = j + 8 * h;
        L[r * 32 + n]      = f32_bf16(p0[j]);
        L[r * 32 + 16 + n] = f32_bf16(p1[j]);
      }
    }
    asm volatile("s_wait_dscnt 0" ::: "memory");        // wave-local LDS RAW
    v16bf pfrag[2];
    #pragma unroll
    for (int mi = 0; mi < 2; ++mi) {
      FragU pf;
      const v4u_ma* ls = (const v4u_ma*)plds[wave][mi];
      pf.u[0] = ls[m * 4 + 2 * h];
      pf.u[1] = ls[m * 4 + 2 * h + 1];
      pfrag[mi] = pf.h;
    }
    #pragma unroll
    for (int t = 0; t < 4; ++t) {
      const v16bf vf = load_frag(buf + (4 + t) * 256, lane);
      o[0][t] = wmma_bf16(pfrag[0], vf, o[0][t]);
      o[1][t] = wmma_bf16(pfrag[1], vf, o[1][t]);
    }
  }
  #pragma unroll
  for (int mi = 0; mi < 2; ++mi) {
    float* Orow = Oact + ((size_t)ba * SEQ + (mt0 + mi) * 16) * DA;
    #pragma unroll
    for (int j = 0; j < 8; ++j) {
      const float inv = 1.f / lrun[mi][j];
      const int r = j + 8 * h;
      #pragma unroll
      for (int t = 0; t < 4; ++t)
        Orow[(size_t)r * DA + t * 16 + n] = o[mi][t][j] * inv;
    }
  }
}

// ---------------- output projection, 2x2 register-blocked ----------------
__global__ __launch_bounds__(256) void final_gemm_kernel(
    const unsigned int* __restrict__ Op, const unsigned int* __restrict__ WoP,
    const float* __restrict__ bo, float* __restrict__ out) {
  const int lane = threadIdx.x & 31, wave = threadIdx.x >> 5;
  const int b = blockIdx.y;
  const int t = blockIdx.x * 8 + wave;                  // 0..2047
  const int mt2 = t >> 5, nt2 = t & 31;                 // 64 x 32 blocks of 32x32
  const unsigned int* A0 = Op + ((size_t)(b * 128 + 2 * mt2 + 0) * 8) * 256;
  const unsigned int* A1 = Op + ((size_t)(b * 128 + 2 * mt2 + 1) * 8) * 256;
  v8f acc[2][2];
  #pragma unroll
  for (int i = 0; i < 2; ++i)
    #pragma unroll
    for (int jj = 0; jj < 2; ++jj) acc[i][jj] = v8f_zero();
  #pragma unroll
  for (int kt = 0; kt < 8; ++kt) {
    const v16bf a0 = load_frag(A0 + (size_t)kt * 256, lane);
    const v16bf a1 = load_frag(A1 + (size_t)kt * 256, lane);
    const v16bf b0 = load_frag(WoP + ((size_t)kt * 64 + 2 * nt2 + 0) * 256, lane);
    const v16bf b1 = load_frag(WoP + ((size_t)kt * 64 + 2 * nt2 + 1) * 256, lane);
    acc[0][0] = wmma_bf16(a0, b0, acc[0][0]);
    acc[0][1] = wmma_bf16(a0, b1, acc[0][1]);
    acc[1][0] = wmma_bf16(a1, b0, acc[1][0]);
    acc[1][1] = wmma_bf16(a1, b1, acc[1][1]);
  }
  const int n = lane & 15, h = lane >> 4;
  #pragma unroll
  for (int i = 0; i < 2; ++i) {
    #pragma unroll
    for (int jj = 0; jj < 2; ++jj) {
      const int nt = 2 * nt2 + jj;
      const float bias = bo[nt * 16 + n];
      #pragma unroll
      for (int j = 0; j < 8; ++j) {
        const int r = (2 * mt2 + i) * 16 + j + 8 * h;
        out[((size_t)b * SEQ + r) * DM + nt * 16 + n] = acc[i][jj][j] + bias;
      }
    }
  }
}

// ---------------- host launcher ----------------
extern "C" void kernel_launch(void* const* d_in, const int* in_sizes, int n_in,
                              void* d_out, int out_size, void* d_ws, size_t ws_size,
                              hipStream_t stream) {
  (void)in_sizes; (void)n_in; (void)out_size; (void)ws_size;
  const float* x  = (const float*)d_in[0];
  const float* Wq = (const float*)d_in[1];
  const float* bq = (const float*)d_in[2];
  const float* Wk = (const float*)d_in[3];
  const float* bk = (const float*)d_in[4];
  const float* Wv = (const float*)d_in[5];
  const float* bv = (const float*)d_in[6];
  const float* Wr = (const float*)d_in[7];
  const float* br = (const float*)d_in[8];
  const float* Wo = (const float*)d_in[9];
  const float* bo = (const float*)d_in[10];
  float* out = (float*)d_out;

  char* ws = (char*)d_ws;
  int* idx            = (int*)(ws + OFF_IDX);
  unsigned int* Xp    = (unsigned int*)(ws + OFF_XP);
  unsigned int* Wp    = (unsigned int*)(ws + OFF_WP);
  float* Qf           = (float*)(ws + OFF_QF);
  float* Kf           = (float*)(ws + OFF_KF);
  float* Vf           = (float*)(ws + OFF_VF);
  unsigned int* Qp    = (unsigned int*)(ws + OFF_QP);
  unsigned int* Kp    = (unsigned int*)(ws + OFF_KP);
  unsigned int* Vp    = (unsigned int*)(ws + OFF_VP);
  float* Oact         = (float*)(ws + OFF_OACT);
  unsigned int* Op    = (unsigned int*)(ws + OFF_OP);
  unsigned int* WoP   = (unsigned int*)(ws + OFF_WOP);

  router_kernel    <<<dim3(BATCH),      dim3(256), 0, stream>>>(x, Wr, br, idx);
  pack_x_kernel    <<<dim3(16384),      dim3(256), 0, stream>>>(x, Xp);
  pack_w_kernel    <<<dim3(6144),       dim3(256), 0, stream>>>(Wq, Wk, Wv, idx, Wp);
  proj_gemm_kernel <<<dim3(16, 48),     dim3(256), 0, stream>>>(Xp, Wp, idx, bq, bk, bv, Qf, Kf, Vf);
  repack_q_kernel  <<<dim3(4096),       dim3(256), 0, stream>>>(Qf, Qp);
  repack_k_kernel  <<<dim3(4096),       dim3(256), 0, stream>>>(Kf, Kp);
  repack_v_kernel  <<<dim3(4096),       dim3(256), 0, stream>>>(Vf, Vp);
  attn_kernel      <<<dim3(8, 16),      dim3(256), 0, stream>>>(Qp, Kp, Vp, Oact);
  repack_o_kernel  <<<dim3(4096),       dim3(256), 0, stream>>>(Oact, Op);
  pack_wo_kernel   <<<dim3(512),        dim3(256), 0, stream>>>(Wo, WoP);
  final_gemm_kernel<<<dim3(256, BATCH), dim3(256), 0, stream>>>(Op, WoP, bo, out);
}
